// RelativePositionalSelfAttention_11570641895988
// MI455X (gfx1250) — matrix operand order repaired and checked by reading, not compiled, not run on hardware
//
#include <hip/hip_runtime.h>

// CDNA5 / gfx1250 relative-positional self-attention using v_wmma_f32_16x16x32_bf16.
// One workgroup per (b,h) head: B*H = 4096 blocks, 256 threads (8 wave32 waves).

typedef __attribute__((ext_vector_type(16))) __bf16 v16bf;
typedef __attribute__((ext_vector_type(8)))  __bf16 v8bf;   // 16 B, align 16
typedef __attribute__((ext_vector_type(4)))  __bf16 v4bf;   // 8 B, align 8
typedef __attribute__((ext_vector_type(8)))  float  v8f;

#define S_LEN 81
#define D_DIM 64
#define SPAD  96      // 6 tiles of 16 rows/cols
#define EROWS 163     // 2*S+1 relative positions
#define EPAD  176     // 11 tiles of 16
#define QSTR  72      // bf16 row stride (144 B, 16B-aligned rows)
#define VTSTR 104     // bf16 row stride for transposed V (208 B)
#define PSTR  184     // bf16 row stride for P / probs (368 B)
#define SSTR  100     // f32 row stride for raw scores
#define NWAVE 8
#define SCALE 0.03608439182435161f   // 1/sqrt(768) per reference

#define SMEM_BYTES (2*SPAD*QSTR*2 /*Q,K*/ + D_DIM*VTSTR*2 /*Vt*/ + EPAD*QSTR*2 /*emb*/ \
                    + SPAD*PSTR*2 /*P/probs bf16*/ + SPAD*SSTR*4 /*scores f32*/)

#define WMMA_BF16(a, b, c) \
  __builtin_amdgcn_wmma_f32_16x16x32_bf16(false, (a), false, (b), (short)0, (c), false, false)

#define CAT16(lo, hi) __builtin_shufflevector((lo), (hi), 0,1,2,3,4,5,6,7,8,9,10,11,12,13,14,15)

// A fragment: 16x32 bf16 tile, A layout (ISA 7.12.2):
// lane<16: K = {0..7,16..23}, lane>=16: K = {8..15,24..31}; m = lane&15.
// Two aligned 16B runs per lane -> 2x ds_load_b128.
__device__ inline v16bf load_a_frag(const __bf16* base, int stride, int row0, int k0, int lane) {
  const int m  = lane & 15;
  const int kh = (lane >> 4) << 3;
  const __bf16* p = base + (row0 + m) * stride + k0 + kh;
  const v8bf lo = *(const v8bf*)(p);        // K = kh+0..7
  const v8bf hi = *(const v8bf*)(p + 16);   // K = kh+16..23
  return CAT16(lo, hi);
}

// B fragment for B = Xᵀ (X row-major, B[k][n] = X[row0+n][k0+k]):
// n = lane&15, lane<16: K = {0..7,16..23}, lane>=16: K = {8..15,24..31} -> 2x ds_load_b128.
__device__ inline v16bf load_bt_frag(const __bf16* base, int stride, int row0, int k0, int lane) {
  const int n  = lane & 15;
  const int kh = (lane >> 4) << 3;
  const __bf16* p = base + (row0 + n) * stride + k0 + kh;
  const v8bf lo = *(const v8bf*)(p);
  const v8bf hi = *(const v8bf*)(p + 16);
  return CAT16(lo, hi);
}

__device__ inline void store_bf16x4(__bf16* dst, float4 t) {
  v4bf v;
  v[0] = (__bf16)t.x; v[1] = (__bf16)t.y; v[2] = (__bf16)t.z; v[3] = (__bf16)t.w;
  *(v4bf*)dst = v;   // single ds_store_b64
}

__global__ void __launch_bounds__(256, 1)
rel_attn_kernel(const float* __restrict__ Qg, const float* __restrict__ Kg,
                const float* __restrict__ Vg, const float* __restrict__ Eg,
                const int* __restrict__ use_causal, float* __restrict__ out) {
  extern __shared__ __align__(16) char smem[];
  __bf16* Qs = (__bf16*)smem;               // [SPAD][QSTR]
  __bf16* Ks = Qs + SPAD * QSTR;            // [SPAD][QSTR]
  __bf16* Vt = Ks + SPAD * QSTR;            // [D_DIM][VTSTR]   V transposed: Vt[d][j]
  __bf16* Es = Vt + D_DIM * VTSTR;          // [EPAD][QSTR]
  __bf16* Ps = Es + EPAD * QSTR;            // [SPAD][PSTR]  P = Q@embᵀ, then probs
  float*  Ss = (float*)(Ps + SPAD * PSTR);  // [SPAD][SSTR]  raw QKᵀ scores

  const int tid  = threadIdx.x;
  const int wid  = tid >> 5;
  const int lane = tid & 31;
  const int bh   = blockIdx.x;
  const int causal = use_causal[0];

  const float4* q4 = (const float4*)(Qg + (size_t)bh * S_LEN * D_DIM);
  const float4* k4 = (const float4*)(Kg + (size_t)bh * S_LEN * D_DIM);
  const float4* v4 = (const float4*)(Vg + (size_t)bh * S_LEN * D_DIM);
  const float4* e4 = (const float4*)Eg;

  // ---- Phase 0: global fp32 -> LDS bf16 (V stored transposed) ----
  for (int idx = tid; idx < S_LEN * D_DIM / 4; idx += 256) {
    const int r = idx >> 4;            // 16 float4 per 64-wide row
    const int c = (idx & 15) << 2;
    const float4 tq = q4[idx], tk = k4[idx], tv = v4[idx];
    store_bf16x4(Qs + r * QSTR + c, tq);
    store_bf16x4(Ks + r * QSTR + c, tk);
    Vt[(c + 0) * VTSTR + r] = (__bf16)tv.x;   // transpose: Vt[d][j] = V[j][d]
    Vt[(c + 1) * VTSTR + r] = (__bf16)tv.y;
    Vt[(c + 2) * VTSTR + r] = (__bf16)tv.z;
    Vt[(c + 3) * VTSTR + r] = (__bf16)tv.w;
  }
  for (int idx = tid; idx < EROWS * D_DIM / 4; idx += 256) {
    const int r = idx >> 4;
    const int c = (idx & 15) << 2;
    store_bf16x4(Es + r * QSTR + c, e4[idx]);
  }
  // zero-pad: rows 81..95 of Q/K, cols 81..95 of Vt, rows 163..175 of E
  const float4 z4 = make_float4(0.f, 0.f, 0.f, 0.f);
  for (int idx = tid; idx < (SPAD - S_LEN) * (D_DIM / 4); idx += 256) {
    const int r = S_LEN + (idx >> 4);
    const int c = (idx & 15) << 2;
    store_bf16x4(Qs + r * QSTR + c, z4);
    store_bf16x4(Ks + r * QSTR + c, z4);
  }
  for (int idx = tid; idx < D_DIM * 16; idx += 256) {
    const int d = idx >> 4;
    const int j = 80 + (idx & 15);
    if (j >= S_LEN) Vt[d * VTSTR + j] = (__bf16)0.0f;
  }
  for (int idx = tid; idx < (EPAD - EROWS) * (D_DIM / 4); idx += 256) {
    const int r = EROWS + (idx >> 4);
    const int c = (idx & 15) << 2;
    store_bf16x4(Es + r * QSTR + c, z4);
  }
  __syncthreads();

  // ---- Phase 1: scores = Q Kᵀ (6x6 tiles) and P = Q embᵀ (6x11 tiles) ----
  for (int t = wid; t < 36 + 66; t += NWAVE) {
    v8f acc = {};
    if (t < 36) {
      const int ti = t / 6, tj = t % 6;
#pragma unroll
      for (int ks = 0; ks < 2; ++ks) {
        v16bf a = load_a_frag(Qs, QSTR, ti * 16, ks * 32, lane);
        v16bf b = load_bt_frag(Ks, QSTR, tj * 16, ks * 32, lane);
        acc = WMMA_BF16(a, b, acc);
      }
      const int m0 = ti * 16 + ((lane >> 4) << 3);
      const int n  = tj * 16 + (lane & 15);
#pragma unroll
      for (int r = 0; r < 8; ++r) Ss[(m0 + r) * SSTR + n] = acc[r];
    } else {
      const int u = t - 36, ti = u / 11, tr = u % 11;
#pragma unroll
      for (int ks = 0; ks < 2; ++ks) {
        v16bf a = load_a_frag(Qs, QSTR, ti * 16, ks * 32, lane);
        v16bf b = load_bt_frag(Es, QSTR, tr * 16, ks * 32, lane);
        acc = WMMA_BF16(a, b, acc);
      }
      const int m0 = ti * 16 + ((lane >> 4) << 3);
      const int n  = tr * 16 + (lane & 15);
#pragma unroll
      for (int r = 0; r < 8; ++r) Ps[(m0 + r) * PSTR + n] = (__bf16)acc[r];
    }
  }
  __syncthreads();

  // ---- Phase 2: bias gather + scale + mask + softmax (one wave per row) ----
  // Probs are written back into Ps as bf16 (WMMA-A layout source for phase 3).
  // Safe: each row is owned by one wave; all Ps bias reads precede (in lockstep
  // program order) the Ps prob writes for that row.
  for (int row = wid; row < S_LEN; row += NWAVE) {
    const int baseRel = S_LEN - 1 + row;   // rel = S-1+row-j in [row, row+80]
    float vals[3];
    float mx = -3.0e38f;
#pragma unroll
    for (int c = 0; c < 3; ++c) {
      const int j = lane + 32 * c;
      float val = -__builtin_inff();
      if (j < S_LEN && (!causal || j <= row)) {
        const float bias = (float)Ps[row * PSTR + (baseRel - j)];
        val = (Ss[row * SSTR + j] + bias) * SCALE;
      }
      vals[c] = val;
      mx = fmaxf(mx, val);
    }
#pragma unroll
    for (int off = 16; off > 0; off >>= 1) mx = fmaxf(mx, __shfl_xor(mx, off, 32));
    float sum = 0.0f;
#pragma unroll
    for (int c = 0; c < 3; ++c) {
      const float e = __expf(vals[c] - mx);   // exp(-inf - mx) == 0 for masked j
      vals[c] = e;
      sum += e;
    }
#pragma unroll
    for (int off = 16; off > 0; off >>= 1) sum += __shfl_xor(sum, off, 32);
    const float inv = 1.0f / sum;
#pragma unroll
    for (int c = 0; c < 3; ++c)
      Ps[row * PSTR + lane + 32 * c] = (__bf16)(vals[c] * inv);  // cols 81..95 get 0
  }
  __syncthreads();

  // ---- Phase 3: out = attn @ V  (6 row tiles x 4 col tiles, K = 96) ----
  // B[k=j][n=d] = V[j][d] = Vt[d][j]  -> contiguous transposed loader.
  for (int t = wid; t < 24; t += NWAVE) {
    const int ti = t >> 2, td = t & 3;
    v8f acc = {};
#pragma unroll
    for (int ks = 0; ks < 3; ++ks) {
      v16bf a = load_a_frag(Ps, PSTR, ti * 16, ks * 32, lane);
      v16bf b = load_bt_frag(Vt, VTSTR, td * 16, ks * 32, lane);
      acc = WMMA_BF16(a, b, acc);
    }
    const int m0 = ti * 16 + ((lane >> 4) << 3);
    const int n  = td * 16 + (lane & 15);
#pragma unroll
    for (int r = 0; r < 8; ++r) {
      const int m = m0 + r;
      if (m < S_LEN)
        out[((size_t)bh * S_LEN + m) * D_DIM + n] = acc[r];
    }
  }
}

extern "C" void kernel_launch(void* const* d_in, const int* in_sizes, int n_in,
                              void* d_out, int out_size, void* d_ws, size_t ws_size,
                              hipStream_t stream) {
  const float* Q = (const float*)d_in[0];
  const float* K = (const float*)d_in[1];
  const float* V = (const float*)d_in[2];
  const float* E = (const float*)d_in[3];
  const int* use_causal = (const int*)d_in[4];
  float* out = (float*)d_out;

  (void)in_sizes; (void)n_in; (void)out_size; (void)d_ws; (void)ws_size;

  // 4096 heads (B*H), dynamic LDS ~137 KB (within 320 KB WGP budget).
  (void)hipFuncSetAttribute((const void*)rel_attn_kernel,
                            hipFuncAttributeMaxDynamicSharedMemorySize,
                            (int)SMEM_BYTES);
  rel_attn_kernel<<<dim3(256 * 16), dim3(256), SMEM_BYTES, stream>>>(Q, K, V, E, use_causal, out);
}
